// MiniMaxText01Attention_15144054686030
// MI455X (gfx1250) — compile-verified, matrix operand order checked by reading, not run on hardware
//
#include <hip/hip_runtime.h>
#include <hip/hip_bf16.h>

// ---------------------------------------------------------------------------
// MiniMaxText01Attention for MI455X (gfx1250, wave32, WMMA bf16 + TDM)
//   S=2048, HIDDEN=4096, NH=32, NKV=8, HD=128, RD=64, THETA=1e4
// ---------------------------------------------------------------------------

#define SEQ_    2048
#define HID_    4096
#define NH_     32
#define NKV_    8
#define HD_     128
#define QKVW_   6144          // (NH + 2*NKV) * HD
#define GQA_    4             // NH / NKV

typedef __attribute__((ext_vector_type(16))) __bf16 v16bf;
typedef __attribute__((ext_vector_type(8)))  float  v8f;
typedef __attribute__((ext_vector_type(4)))  unsigned int u32x4;
typedef __attribute__((ext_vector_type(8)))  unsigned int u32x8;

union Frag {
  v16bf v;
  unsigned int u[8];
};

__device__ __forceinline__ unsigned short f2bf(float x) {
  __bf16 b = (__bf16)x;
  return __builtin_bit_cast(unsigned short, b);
}

// Load a 16-lane 16x32 (MxK) bf16 fragment in the ISA A/B layout:
//   lane&15 selects the row (A) / column (B); hi = lane>>4.
//   j2 0..3 -> K = hi*8 + 2*j2 ; j2 4..7 -> K = 16 + hi*8 + 2*(j2-4)
// `slice` points at the 32-element contiguous K-slice for this lane's row.
__device__ __forceinline__ void load_frag32(Frag& f, const unsigned short* slice, int hi) {
  const unsigned int* q0 = (const unsigned int*)(slice + hi * 8);
  const unsigned int* q1 = (const unsigned int*)(slice + hi * 8 + 16);
  f.u[0] = q0[0]; f.u[1] = q0[1]; f.u[2] = q0[2]; f.u[3] = q0[3];
  f.u[4] = q1[0]; f.u[5] = q1[1]; f.u[6] = q1[2]; f.u[7] = q1[3];
}

// Same layout, gathering a column from a row-major [K][pitch] LDS tile
// (used for V, stored untransposed). col = &tile[0][n].
__device__ __forceinline__ void load_fragB_gather(Frag& f, const unsigned short* col,
                                                  int kBase, int hi, int pitch) {
#pragma unroll
  for (int j2 = 0; j2 < 8; ++j2) {
    int k = kBase + ((j2 < 4) ? (hi * 8 + 2 * j2) : (16 + hi * 8 + 2 * (j2 - 4)));
    unsigned int lo = col[k * pitch];
    unsigned int hh = col[(k + 1) * pitch];
    f.u[j2] = lo | (hh << 16);
  }
}

// ---------------------------------------------------------------------------
// Tensor Data Mover: async-load a 2D tile of 16-bit elements into LDS.
// D# built per CDNA5 ISA (groups 0 and 1; VADDR2/3
// NULL -> 2D). data_size code 1 = 2 bytes. Tracked by TENSORcnt.
//   tile_d0 / tensor_d0 : elements along the contiguous dim
//   tile_d1 / tensor_d1 : rows
//   stride0             : row stride in elements
// ---------------------------------------------------------------------------
__device__ __forceinline__ void tdm_load_2d_b16(unsigned lds_off, const void* gptr,
                                                unsigned tile_d0, unsigned tile_d1,
                                                unsigned tensor_d0, unsigned tensor_d1,
                                                unsigned stride0) {
  unsigned long long ga = (unsigned long long)(uintptr_t)gptr;
  u32x4 g0;
  g0[0] = 1u;                                    // count=1, user mode
  g0[1] = lds_off;                               // lds_addr (bytes)
  g0[2] = (unsigned)ga;                          // global_addr[31:0]
  g0[3] = (unsigned)(ga >> 32) | (2u << 30);     // global_addr[56:32] | type=2
  u32x8 g1;
  g1[0] = 0x00010000u;                           // workgroup_mask=0, data_size=1 (2B)
  g1[1] = (tensor_d0 & 0xffffu) << 16;           // atomic_barrier=0 | tensor_dim0 lo
  g1[2] = ((tensor_d0 >> 16) & 0xffffu) | ((tensor_d1 & 0xffffu) << 16);
  g1[3] = ((tensor_d1 >> 16) & 0xffffu) | (tile_d0 << 16);   // | tile_dim0
  g1[4] = tile_d1;                               // tile_dim1; tile_dim2=0
  g1[5] = stride0;                               // tensor_dim0_stride lo
  g1[6] = 0u;                                    // stride hi | dim1_stride lo (unused, 2D)
  g1[7] = 0u;
  asm volatile("tensor_load_to_lds %0, %1" :: "s"(g0), "s"(g1) : "memory");
}

// ---------------------------------------------------------------------------
// Generic C(f32) = A @ B GEMM.  A: row-major [M][K] (f32 or bf16-bits),
// B: row-major f32 [K][N].  On-the-fly bf16 conversion into LDS, WMMA bf16.
// Block: 256 threads = 8 waves; tile 128(M) x 64(N), BK=32.
// Wave w owns a 32x32 C tile: rows 32*(w&3), cols 32*(w>>2) -> 2x2 accums.
// ---------------------------------------------------------------------------
template <int A_BF16>
__global__ __launch_bounds__(256) void gemm_wmma(const void* __restrict__ Ap,
                                                 const float* __restrict__ B,
                                                 float* __restrict__ C,
                                                 int M, int N, int K) {
  const int tid  = threadIdx.x;
  const int wave = tid >> 5;
  const int lane = tid & 31;
  const int ln   = lane & 15;
  const int hi   = lane >> 4;
  const int wm   = wave & 3;
  const int wn   = wave >> 2;
  const int rowBase = blockIdx.y * 128;
  const int colBase = blockIdx.x * 64;

  __shared__ unsigned short As[128][32];   // [m][k] bf16 bits (8 KB)
  __shared__ unsigned short Bs[64][32];    // transposed: [n][k] bf16 bits (4 KB)

  v8f acc[2][2] = {};

  const int am = tid >> 1;          // 0..127
  const int ak = (tid & 1) * 16;    // 0,16
  const int bk = tid >> 3;          // 0..31
  const int bn = (tid & 7) * 8;     // 0..56

  for (int k0 = 0; k0 < K; k0 += 32) {
    // ---- stage A tile (128x32) ----
    if (A_BF16) {
      const unsigned short* src =
          (const unsigned short*)Ap + (size_t)(rowBase + am) * K + k0 + ak;
      *(uint4*)&As[am][ak]     = *(const uint4*)src;
      *(uint4*)&As[am][ak + 8] = *(const uint4*)(src + 8);
    } else {
      const float* src = (const float*)Ap + (size_t)(rowBase + am) * K + k0 + ak;
      float4 f0 = *(const float4*)src;
      float4 f1 = *(const float4*)(src + 4);
      float4 f2 = *(const float4*)(src + 8);
      float4 f3 = *(const float4*)(src + 12);
      uint4 pk0, pk1;
      pk0.x = f2bf(f0.x) | ((unsigned)f2bf(f0.y) << 16);
      pk0.y = f2bf(f0.z) | ((unsigned)f2bf(f0.w) << 16);
      pk0.z = f2bf(f1.x) | ((unsigned)f2bf(f1.y) << 16);
      pk0.w = f2bf(f1.z) | ((unsigned)f2bf(f1.w) << 16);
      pk1.x = f2bf(f2.x) | ((unsigned)f2bf(f2.y) << 16);
      pk1.y = f2bf(f2.z) | ((unsigned)f2bf(f2.w) << 16);
      pk1.z = f2bf(f3.x) | ((unsigned)f2bf(f3.y) << 16);
      pk1.w = f2bf(f3.z) | ((unsigned)f2bf(f3.w) << 16);
      *(uint4*)&As[am][ak]     = pk0;
      *(uint4*)&As[am][ak + 8] = pk1;
    }
    // ---- stage B tile (32x64, transposed into [n][k]) ----
    {
      const float* src = B + (size_t)(k0 + bk) * N + colBase + bn;
      float4 f0 = *(const float4*)src;
      float4 f1 = *(const float4*)(src + 4);
      Bs[bn + 0][bk] = f2bf(f0.x);
      Bs[bn + 1][bk] = f2bf(f0.y);
      Bs[bn + 2][bk] = f2bf(f0.z);
      Bs[bn + 3][bk] = f2bf(f0.w);
      Bs[bn + 4][bk] = f2bf(f1.x);
      Bs[bn + 5][bk] = f2bf(f1.y);
      Bs[bn + 6][bk] = f2bf(f1.z);
      Bs[bn + 7][bk] = f2bf(f1.w);
      if (k0 + 32 < K)
        __builtin_prefetch(src + 32 * (size_t)N, 0, 1);
    }
    __syncthreads();

    Frag a0, a1, b0, b1;
    load_frag32(a0, &As[wm * 32 + ln][0],      hi);
    load_frag32(a1, &As[wm * 32 + 16 + ln][0], hi);
    load_frag32(b0, &Bs[wn * 32 + ln][0],      hi);
    load_frag32(b1, &Bs[wn * 32 + 16 + ln][0], hi);
    acc[0][0] = __builtin_amdgcn_wmma_f32_16x16x32_bf16(false, a0.v, false, b0.v,
                                                        (short)0, acc[0][0], false, false);
    acc[0][1] = __builtin_amdgcn_wmma_f32_16x16x32_bf16(false, a0.v, false, b1.v,
                                                        (short)0, acc[0][1], false, false);
    acc[1][0] = __builtin_amdgcn_wmma_f32_16x16x32_bf16(false, a1.v, false, b0.v,
                                                        (short)0, acc[1][0], false, false);
    acc[1][1] = __builtin_amdgcn_wmma_f32_16x16x32_bf16(false, a1.v, false, b1.v,
                                                        (short)0, acc[1][1], false, false);
    __syncthreads();
  }

  // epilogue: C-layout is M = r + 8*hi, N = ln
#pragma unroll
  for (int mi = 0; mi < 2; ++mi)
#pragma unroll
    for (int r = 0; r < 8; ++r) {
      size_t row = (size_t)(rowBase + wm * 32 + mi * 16 + r + 8 * hi);
      float* crow = C + row * N + colBase + wn * 32;
      crow[ln]      = acc[mi][0][r];
      crow[16 + ln] = acc[mi][1][r];
    }
}

// ---------------------------------------------------------------------------
// RoPE on q/k (first RD=64 dims per head, half=32) + split qkv -> bf16 q,k,v.
// ---------------------------------------------------------------------------
__global__ __launch_bounds__(256) void rope_split(const float* __restrict__ qkv,
                                                  const int* __restrict__ positions,
                                                  unsigned short* __restrict__ qb,
                                                  unsigned short* __restrict__ kb,
                                                  unsigned short* __restrict__ vb) {
  const int s = blockIdx.x;
  const float pos = (float)positions[s];
  const float* row = qkv + (size_t)s * QKVW_;
  const float LOG_THETA_OVER_HALF = 9.210340371976184f / 32.0f;   // ln(1e4)/32

  for (int j = threadIdx.x; j < QKVW_; j += 256) {
    float x = row[j];
    int seg, off;
    if (j < NH_ * HD_)                   { seg = 0; off = j; }
    else if (j < (NH_ + NKV_) * HD_)     { seg = 1; off = j - NH_ * HD_; }
    else                                 { seg = 2; off = j - (NH_ + NKV_) * HD_; }
    const int d = off & (HD_ - 1);
    float out = x;
    if (seg < 2 && d < 64) {
      const int i = d & 31;
      const float inv_freq = __expf(-(float)i * LOG_THETA_OVER_HALF);
      const float fr = pos * inv_freq;
      float sn, c;
      __sincosf(fr, &sn, &c);
      const float partner = row[j + ((d < 32) ? 32 : -32)];
      out = (d < 32) ? (x * c - partner * sn)    // o1 = x1*cos - x2*sin
                     : (x * c + partner * sn);   // o2 = x2*cos + x1*sin
    }
    const unsigned short b = f2bf(out);
    if (seg == 0)      qb[(size_t)s * (NH_ * HD_) + off] = b;
    else if (seg == 1) kb[(size_t)s * (NKV_ * HD_) + off] = b;
    else               vb[(size_t)s * (NKV_ * HD_) + off] = b;
  }
}

// ---------------------------------------------------------------------------
// Flash-style causal GQA attention.
// Block: 128 threads = 4 waves. blockIdx.x = head, blockIdx.y = 64-row q tile.
// Each wave owns 16 query rows and a 16x128 f32 O accumulator (8 v8f frags).
// K/V tiles are DMA'd into LDS by the Tensor Data Mover (wave 0 issues the
// tensor_load_to_lds pair, waits on TENSORcnt, then the WG barrier releases
// everyone). S = Q K^T via WMMA; online softmax with __shfl_xor row
// reductions; P -> LDS (layout conversion) -> PV WMMA.
// ---------------------------------------------------------------------------
__global__ __launch_bounds__(128) void attn_kernel(const unsigned short* __restrict__ qbuf,
                                                   const unsigned short* __restrict__ kbuf,
                                                   const unsigned short* __restrict__ vbuf,
                                                   unsigned short* __restrict__ attnb) {
  const int h     = blockIdx.x;         // 0..31
  const int qtile = blockIdx.y;         // 0..31
  const int kvh   = h / GQA_;
  const int tid   = threadIdx.x;
  const int w     = tid >> 5;           // wave 0..3
  const int lane  = tid & 31;
  const int ln    = lane & 15;
  const int hi    = lane >> 4;
  const int qRow0 = qtile * 64 + w * 16;
  const float scale = 0.08838834764831845f;   // HD^-0.5

  __shared__ unsigned short Ks[64][HD_];      // 16 KB
  __shared__ unsigned short Vs[64][HD_];      // 16 KB
  __shared__ unsigned short Pt[4][16][64];    // 8 KB, per-wave P tiles

  // preload Q fragments (A layout; lane ln -> query row qRow0+ln)
  Frag qf[4];
  {
    const unsigned short* qrow =
        qbuf + (size_t)(qRow0 + ln) * (NH_ * HD_) + h * HD_;
#pragma unroll
    for (int d4 = 0; d4 < 4; ++d4) load_frag32(qf[d4], qrow + d4 * 32, hi);
  }

  v8f acc[8] = {};
  float mrow[8], lrow[8];
#pragma unroll
  for (int r = 0; r < 8; ++r) { mrow[r] = -1e30f; lrow[r] = 0.0f; }

  for (int kt = 0; kt <= qtile; ++kt) {
    const int t0 = kt * 64;

    // ---- TDM async stage of K and V tiles (64 x 128 bf16 each) ----
    if (w == 0) {
      const unsigned ldsK = (unsigned)(uintptr_t)&Ks[0][0];
      const unsigned ldsV = (unsigned)(uintptr_t)&Vs[0][0];
      const size_t goff = (size_t)t0 * (NKV_ * HD_) + kvh * HD_;
      tdm_load_2d_b16(ldsK, kbuf + goff, HD_, 64, HD_, 64, NKV_ * HD_);
      tdm_load_2d_b16(ldsV, vbuf + goff, HD_, 64, HD_, 64, NKV_ * HD_);
      __builtin_amdgcn_s_wait_tensorcnt(0);
    }
    __syncthreads();

    // ---- S = Q @ K^T  (4 key sub-tiles x 4 d-steps) ----
    v8f sc[4] = {};
#pragma unroll
    for (int nt = 0; nt < 4; ++nt) {
#pragma unroll
      for (int d4 = 0; d4 < 4; ++d4) {
        Frag kf;
        load_frag32(kf, &Ks[nt * 16 + ln][d4 * 32], hi);
        sc[nt] = __builtin_amdgcn_wmma_f32_16x16x32_bf16(
            false, qf[d4].v, false, kf.v, (short)0, sc[nt], false, false);
      }
    }

    // ---- causal mask + scale; row max (C-layout: M=r+8*hi, N=ln) ----
    float mnew[8];
#pragma unroll
    for (int r = 0; r < 8; ++r) mnew[r] = mrow[r];
#pragma unroll
    for (int nt = 0; nt < 4; ++nt) {
      const int tcol = t0 + nt * 16 + ln;
#pragma unroll
      for (int r = 0; r < 8; ++r) {
        const int qr = qRow0 + r + 8 * hi;
        float v = sc[nt][r] * scale;
        if (tcol > qr) v = -1e30f;
        sc[nt][r] = v;
        mnew[r] = fmaxf(mnew[r], v);
      }
    }
#pragma unroll
    for (int m = 1; m < 16; m <<= 1)
#pragma unroll
      for (int r = 0; r < 8; ++r)
        mnew[r] = fmaxf(mnew[r], __shfl_xor(mnew[r], m, 32));

    // ---- rescale O, exponentiate, stash P into LDS (A layout rows) ----
    float alpha[8];
#pragma unroll
    for (int r = 0; r < 8; ++r) {
      alpha[r] = __expf(mrow[r] - mnew[r]);
      mrow[r]  = mnew[r];
    }
#pragma unroll
    for (int dt = 0; dt < 8; ++dt)
#pragma unroll
      for (int r = 0; r < 8; ++r) acc[dt][r] *= alpha[r];

    float lsum[8] = {0.f, 0.f, 0.f, 0.f, 0.f, 0.f, 0.f, 0.f};
#pragma unroll
    for (int nt = 0; nt < 4; ++nt)
#pragma unroll
      for (int r = 0; r < 8; ++r) {
        const float p = __expf(sc[nt][r] - mnew[r]);
        lsum[r] += p;
        Pt[w][r + 8 * hi][nt * 16 + ln] = f2bf(p);
      }
#pragma unroll
    for (int m = 1; m < 16; m <<= 1)
#pragma unroll
      for (int r = 0; r < 8; ++r) lsum[r] += __shfl_xor(lsum[r], m, 32);
#pragma unroll
    for (int r = 0; r < 8; ++r) lrow[r] = lrow[r] * alpha[r] + lsum[r];

    __syncthreads();   // P visible across lanes

    // ---- O += P @ V  (2 t-steps x 8 d sub-tiles) ----
    Frag pf[2];
    load_frag32(pf[0], &Pt[w][ln][0],  hi);
    load_frag32(pf[1], &Pt[w][ln][32], hi);
#pragma unroll
    for (int dt = 0; dt < 8; ++dt) {
#pragma unroll
      for (int ts = 0; ts < 2; ++ts) {
        Frag vf;
        load_fragB_gather(vf, &Vs[0][dt * 16 + ln], ts * 32, hi, HD_);
        acc[dt] = __builtin_amdgcn_wmma_f32_16x16x32_bf16(
            false, pf[ts].v, false, vf.v, (short)0, acc[dt], false, false);
      }
    }
    __syncthreads();   // done with Ks/Vs/Pt before restaging
  }

  // ---- normalize and store O as bf16 [S][NH*HD] ----
  float invl[8];
#pragma unroll
  for (int r = 0; r < 8; ++r) invl[r] = 1.0f / lrow[r];
#pragma unroll
  for (int dt = 0; dt < 8; ++dt)
#pragma unroll
    for (int r = 0; r < 8; ++r) {
      const size_t row = (size_t)(qRow0 + r + 8 * hi);
      attnb[row * (NH_ * HD_) + h * HD_ + dt * 16 + ln] = f2bf(acc[dt][r] * invl[r]);
    }
}

// ---------------------------------------------------------------------------
extern "C" void kernel_launch(void* const* d_in, const int* in_sizes, int n_in,
                              void* d_out, int out_size, void* d_ws, size_t ws_size,
                              hipStream_t stream) {
  const float* hidden = (const float*)d_in[0];   // [2048][4096]
  const float* Wqkv   = (const float*)d_in[1];   // [4096][6144]
  const float* Wo     = (const float*)d_in[2];   // [4096][4096]
  const int*   pos    = (const int*)d_in[3];     // [2048]
  float* out = (float*)d_out;                    // [2048][4096]

  char* ws = (char*)d_ws;
  size_t off = 0;
  float* qkv = (float*)(ws + off);                  off += (size_t)SEQ_ * QKVW_ * 4;       // 48 MB
  unsigned short* qb = (unsigned short*)(ws + off); off += (size_t)SEQ_ * NH_ * HD_ * 2;   // 16 MB
  unsigned short* kb = (unsigned short*)(ws + off); off += (size_t)SEQ_ * NKV_ * HD_ * 2;  // 4 MB
  unsigned short* vb = (unsigned short*)(ws + off); off += (size_t)SEQ_ * NKV_ * HD_ * 2;  // 4 MB
  unsigned short* attnb = (unsigned short*)(ws + off);                                     // 16 MB

  // 1) qkv = hidden @ Wqkv           (f32 in, bf16 WMMA, f32 out)
  gemm_wmma<0><<<dim3(QKVW_ / 64, SEQ_ / 128), 256, 0, stream>>>(
      hidden, Wqkv, qkv, SEQ_, QKVW_, HID_);

  // 2) RoPE + split into bf16 q/k/v
  rope_split<<<dim3(SEQ_), 256, 0, stream>>>(qkv, pos, qb, kb, vb);

  // 3) causal GQA flash attention (TDM-staged K/V) -> bf16 attn
  attn_kernel<<<dim3(NH_, SEQ_ / 64), 128, 0, stream>>>(qb, kb, vb, attnb);

  // 4) out = attn @ Wo               (bf16 A, f32 B, f32 out)
  gemm_wmma<1><<<dim3(HID_ / 64, SEQ_ / 128), 256, 0, stream>>>(
      attnb, Wo, out, SEQ_, HID_, HID_);
}
